// WindowAttention_86217173500430
// MI455X (gfx1250) — matrix operand order repaired and checked by reading, not compiled, not run on hardware
//
#include <hip/hip_runtime.h>

// ---------------- types ----------------
typedef __attribute__((ext_vector_type(16))) __bf16 v16bf;
typedef __attribute__((ext_vector_type(8)))  float  v8f;
typedef __attribute__((ext_vector_type(4)))  unsigned int v4u;

#define DIM    192
#define NH     6
#define HD     32
#define NTOK   64
#define SCALEQ 0.17677669529663687f  // 32^-0.5

#define XPITCH 200   // bf16 elems per row for X/Q/K/O (192 + 8 pad, 16B-aligned rows)
#define VPITCH 72    // token pitch for V^T (64 + 8 pad)
#define PPITCH 72    // key pitch for P    (64 + 8 pad)

// ws layout (bytes)
#define OFF_WQKV  0          // 216 tiles * 1024 B  (bf16 fragment-major)
#define OFF_WPROJ 221184     // 72 tiles * 1024 B
#define OFF_BIAS  294912     // 6*64*64 f32

static __device__ __forceinline__ unsigned short f2bf(float f) {
  union { float f; unsigned u; } v; v.f = f;
  unsigned r = v.u + 0x7FFFu + ((v.u >> 16) & 1u);   // round-to-nearest-even
  return (unsigned short)(r >> 16);
}

// A fragment (16x32 bf16, M x K): lane<16 -> rows M=lane, K = {0..15};
// lane>=16 -> rows M=lane-16, K = {16..31}. Two ds_load_b128 per lane.
static __device__ __forceinline__ v16bf ldsA(const unsigned short* base, int pitch,
                                             int row0, int col0, int lane) {
  const unsigned short* p = base + (row0 + (lane & 15)) * pitch + col0 + ((lane >> 4) << 4);
  union { v16bf v; v4u q[2]; } f;
  f.q[0] = *(const v4u*)(p);
  f.q[1] = *(const v4u*)(p + 8);
  return f.v;
}

// B fragment (32x16 bf16, K x N): lane&15 -> column N, lane>>4 -> K half.
// Source matrix stored so that the 16 K-values for a lane are contiguous.
static __device__ __forceinline__ v16bf ldsB(const unsigned short* base, int pitch,
                                             int row0, int k0, int lane) {
  const unsigned short* p = base + (row0 + (lane & 15)) * pitch + k0 + ((lane >> 4) << 4);
  union { v16bf v; v4u q[2]; } f;
  f.q[0] = *(const v4u*)(p);
  f.q[1] = *(const v4u*)(p + 8);
  return f.v;
}

// Pre-swizzled weight B fragment: 1024 B per tile, 32 B contiguous per lane.
static __device__ __forceinline__ v16bf gfrag(const unsigned short* tile, int lane) {
  const v4u* g = (const v4u*)(tile + lane * 16);
  union { v16bf v; v4u q[2]; } f;
  f.q[0] = g[0]; f.q[1] = g[1];
  return f.v;
}

static __device__ __forceinline__ v8f wmma_bf16(v16bf a, v16bf b, v8f c) {
  return __builtin_amdgcn_wmma_f32_16x16x32_bf16(false, a, false, b, (short)0, c, false, false);
}

// ---------------- prep: weight swizzle (f32 -> bf16 fragments) + bias gather ----------------
__global__ void prep_kernel(const float* __restrict__ w_qkv, const float* __restrict__ w_proj,
                            const float* __restrict__ rpbt,
                            unsigned short* __restrict__ wqkv_f,
                            unsigned short* __restrict__ wproj_f,
                            float* __restrict__ biasTab) {
  const int NQ = 216 * 512, NP = 72 * 512, NB = NH * 64 * 64;
  int gid = blockIdx.x * blockDim.x + threadIdx.x;
  if (gid < NQ) {                       // w_qkv: 36 N-tiles x 6 K-tiles
    int tile = gid >> 9, within = gid & 511;
    int lane = within >> 4, e = within & 15;
    int nt = tile / 6, kt = tile % 6;
    int k = kt * 32 + ((lane >> 4) << 4) + e;
    int n = nt * 16 + (lane & 15);
    wqkv_f[gid] = f2bf(w_qkv[k * 576 + n]);
  } else if (gid < NQ + NP) {           // w_proj: 12 N-tiles x 6 K-tiles
    int id = gid - NQ;
    int tile = id >> 9, within = id & 511;
    int lane = within >> 4, e = within & 15;
    int nt = tile / 6, kt = tile % 6;
    int k = kt * 32 + ((lane >> 4) << 4) + e;
    int n = nt * 16 + (lane & 15);
    wproj_f[id] = f2bf(w_proj[k * 192 + n]);
  } else if (gid < NQ + NP + NB) {      // rel-pos bias gather -> bias[h][m][j]
    int id = gid - NQ - NP;
    int h = id / 4096, m = (id >> 6) & 63, j = id & 63;
    int dw = (m >> 3) - (j >> 3) + 7;
    int dh = (m & 7)  - (j & 7)  + 7;
    biasTab[id] = rpbt[(dw * 15 + dh) * NH + h];
  }
}

// ---------------- fused window attention ----------------
__global__ __launch_bounds__(192)
void swin_attn_kernel(const float* __restrict__ x_in, const float* __restrict__ mask,
                      const float* __restrict__ b_qkv, const float* __restrict__ b_proj,
                      const unsigned short* __restrict__ wqkv_f,
                      const unsigned short* __restrict__ wproj_f,
                      const float* __restrict__ biasTab,
                      float* __restrict__ out) {
  extern __shared__ __align__(16) unsigned char smem[];
  unsigned short* lds_x  = (unsigned short*)smem;            // 64 x XPITCH
  unsigned short* lds_q  = lds_x  + NTOK * XPITCH;           // 64 x XPITCH (pre-scaled)
  unsigned short* lds_k  = lds_q  + NTOK * XPITCH;           // 64 x XPITCH
  unsigned short* lds_vT = lds_k  + NTOK * XPITCH;           // 192 x VPITCH  (dim-major)
  unsigned short* lds_p  = lds_vT + DIM * VPITCH;            // 6 x 64 x PPITCH
  unsigned short* lds_o  = lds_p  + NH * NTOK * PPITCH;      // 64 x XPITCH

  const int tid  = threadIdx.x;
  const int wave = __builtin_amdgcn_readfirstlane(tid >> 5); // scalar wave id
  const int lane = tid & 31;
  const int jcol = lane & 15;
  const int m0off = (lane >> 4) << 3;
  const int b    = blockIdx.x;
  const int widx = b & 63;                                   // window index within image
  const float* xg = x_in + (size_t)b * (NTOK * DIM);

  // warm caches for the L2-resident weight fragments (global_prefetch_b8)
  for (int i = tid; i < 216; i += 192) __builtin_prefetch(wqkv_f + i * 512, 0, 1);
  for (int i = tid; i < 72;  i += 192) __builtin_prefetch(wproj_f + i * 512, 0, 1);

  // ---- stage 0: X f32 -> bf16 LDS ----
  for (int i = tid; i < (NTOK * DIM) / 4; i += 192) {
    int idx = i * 4;
    int tok = idx / DIM, c = idx % DIM;
    const float4 v = *(const float4*)(xg + idx);             // global_load_b128
    unsigned short* d = lds_x + tok * XPITCH + c;
    d[0] = f2bf(v.x); d[1] = f2bf(v.y); d[2] = f2bf(v.z); d[3] = f2bf(v.w);
  }
  __syncthreads();

  // ---- stage 1: QKV = X @ Wqkv + b  (144 tiles, 24 per wave) ----
  for (int t = wave; t < 144; t += 6) {
    const int tS = __builtin_amdgcn_readfirstlane(t);        // tile index is wave-uniform
    const int mt = tS & 3, nt = tS >> 2;
    v8f acc = {0.f, 0.f, 0.f, 0.f, 0.f, 0.f, 0.f, 0.f};
#pragma unroll
    for (int kt = 0; kt < 6; ++kt) {
      v16bf a = ldsA(lds_x, XPITCH, mt * 16, kt * 32, lane);
      v16bf w = gfrag(wqkv_f + (nt * 6 + kt) * 512, lane);
      acc = wmma_bf16(a, w, acc);
    }
    const float bq  = b_qkv[nt * 16 + jcol];
    const int   mat = nt / 12;                               // uniform: 0=q, 1=k, 2=v
    const int   cc  = (nt - mat * 12) * 16 + jcol;           // h*32+d within 0..191
    const int   tok0 = mt * 16 + m0off;
    if (mat == 0) {
#pragma unroll
      for (int r = 0; r < 8; ++r)
        lds_q[(tok0 + r) * XPITCH + cc] = f2bf((acc[r] + bq) * SCALEQ);
    } else if (mat == 1) {
#pragma unroll
      for (int r = 0; r < 8; ++r)
        lds_k[(tok0 + r) * XPITCH + cc] = f2bf(acc[r] + bq);
    } else {
#pragma unroll
      for (int r = 0; r < 8; ++r)
        lds_vT[cc * VPITCH + tok0 + r] = f2bf(acc[r] + bq);  // transposed, contiguous in r
    }
  }
  __syncthreads();

  // ---- stage 2: per-head attention (wave == head) ----
  {
    const int h = wave;                                      // 0..5
    unsigned short* ph   = lds_p + h * NTOK * PPITCH;
    const float* biasH   = biasTab + h * 64 * 64;
    const float* maskW   = mask + widx * 64 * 64;

    for (int mt = 0; mt < 4; ++mt) {
      v16bf aq = ldsA(lds_q, XPITCH, mt * 16, h * 32, lane); // K = head_dim = 32 exactly
      v8f s[4];
#pragma unroll
      for (int nt = 0; nt < 4; ++nt) {
        v16bf bk = ldsB(lds_k, XPITCH, nt * 16, h * 32, lane);
        v8f z = {0.f, 0.f, 0.f, 0.f, 0.f, 0.f, 0.f, 0.f};
        s[nt] = wmma_bf16(aq, bk, z);                        // S = (Q*scale) K^T
      }
      int m0 = mt * 16 + m0off;
#pragma unroll
      for (int nt = 0; nt < 4; ++nt) {
        int j = nt * 16 + jcol;
#pragma unroll
        for (int r = 0; r < 8; ++r)
          s[nt][r] += biasH[(m0 + r) * 64 + j] + maskW[(m0 + r) * 64 + j];
      }
      // softmax per row: row lives in one 16-lane half + 4 nt-tiles
#pragma unroll
      for (int r = 0; r < 8; ++r) {
        float mx = fmaxf(fmaxf(s[0][r], s[1][r]), fmaxf(s[2][r], s[3][r]));
        for (int off = 8; off >= 1; off >>= 1) mx = fmaxf(mx, __shfl_xor(mx, off, 32));
        float e0 = __expf(s[0][r] - mx), e1 = __expf(s[1][r] - mx);
        float e2 = __expf(s[2][r] - mx), e3 = __expf(s[3][r] - mx);
        float sum = (e0 + e1) + (e2 + e3);
        for (int off = 8; off >= 1; off >>= 1) sum += __shfl_xor(sum, off, 32);
        float rinv = __builtin_amdgcn_rcpf(sum);
        s[0][r] = e0 * rinv; s[1][r] = e1 * rinv; s[2][r] = e2 * rinv; s[3][r] = e3 * rinv;
      }
#pragma unroll
      for (int nt = 0; nt < 4; ++nt) {
        int j = nt * 16 + jcol;
#pragma unroll
        for (int r = 0; r < 8; ++r)
          ph[(m0 + r) * PPITCH + j] = f2bf(s[nt][r]);
      }
    }
    // O_h = P @ V_h  (LDS is in-order per wave: P write -> read is safe)
    for (int mt = 0; mt < 4; ++mt) {
#pragma unroll
      for (int nto = 0; nto < 2; ++nto) {
        v8f acc = {0.f, 0.f, 0.f, 0.f, 0.f, 0.f, 0.f, 0.f};
#pragma unroll
        for (int kt = 0; kt < 2; ++kt) {
          v16bf a  = ldsA(ph, PPITCH, mt * 16, kt * 32, lane);
          v16bf bv = ldsB(lds_vT, VPITCH, h * 32 + nto * 16, kt * 32, lane);
          acc = wmma_bf16(a, bv, acc);
        }
        int col  = h * 32 + nto * 16 + jcol;
        int tok0 = mt * 16 + m0off;
#pragma unroll
        for (int r = 0; r < 8; ++r)
          lds_o[(tok0 + r) * XPITCH + col] = f2bf(acc[r]);
      }
    }
  }
  __syncthreads();

  // ---- stage 3: out = O @ Wproj + b  (48 tiles, 8 per wave) ----
  float* og = out + (size_t)b * (NTOK * DIM);
  for (int t = wave; t < 48; t += 6) {
    const int tS = __builtin_amdgcn_readfirstlane(t);
    const int mt = tS & 3, nt = tS >> 2;
    v8f acc = {0.f, 0.f, 0.f, 0.f, 0.f, 0.f, 0.f, 0.f};
#pragma unroll
    for (int kt = 0; kt < 6; ++kt) {
      v16bf a = ldsA(lds_o, XPITCH, mt * 16, kt * 32, lane);
      v16bf w = gfrag(wproj_f + (nt * 6 + kt) * 512, lane);
      acc = wmma_bf16(a, w, acc);
    }
    const float bp  = b_proj[nt * 16 + jcol];
    const int  tok0 = mt * 16 + m0off;
#pragma unroll
    for (int r = 0; r < 8; ++r)
      og[(tok0 + r) * DIM + nt * 16 + jcol] = acc[r] + bp;
  }
}

extern "C" void kernel_launch(void* const* d_in, const int* in_sizes, int n_in,
                              void* d_out, int out_size, void* d_ws, size_t ws_size,
                              hipStream_t stream) {
  const float* inputs = (const float*)d_in[0];
  const float* mask   = (const float*)d_in[1];
  const float* w_qkv  = (const float*)d_in[2];
  const float* b_qkv  = (const float*)d_in[3];
  const float* rpbt   = (const float*)d_in[4];
  const float* w_proj = (const float*)d_in[5];
  const float* b_proj = (const float*)d_in[6];
  float* out = (float*)d_out;

  unsigned char* ws = (unsigned char*)d_ws;
  unsigned short* wqkv_f  = (unsigned short*)(ws + OFF_WQKV);
  unsigned short* wproj_f = (unsigned short*)(ws + OFF_WPROJ);
  float*          biasTab = (float*)(ws + OFF_BIAS);

  const int B = in_sizes[0] / (NTOK * DIM);   // 2048 windows

  // 216*512 + 72*512 + 6*64*64 = 172032 elements -> 672 blocks of 256
  prep_kernel<<<672, 256, 0, stream>>>(w_qkv, w_proj, rpbt, wqkv_f, wproj_f, biasTab);

  const size_t smem = (size_t)(NTOK * XPITCH * 3 + DIM * VPITCH +
                               NH * NTOK * PPITCH + NTOK * XPITCH) * sizeof(unsigned short);
  swin_attn_kernel<<<B, 192, smem, stream>>>(inputs, mask, b_qkv, b_proj,
                                             wqkv_f, wproj_f, biasTab, out);
}
